// GAT_36945308680705
// MI455X (gfx1250) — matrix operand order repaired and checked by reading, not compiled, not run on hardware
//
#include <hip/hip_runtime.h>

typedef __attribute__((ext_vector_type(16))) _Float16 v16h;
typedef __attribute__((ext_vector_type(8)))  _Float16 v8h;
typedef __attribute__((ext_vector_type(4)))  _Float16 v4h;
typedef __attribute__((ext_vector_type(8)))  float    v8f;

// ---------- ordered-uint mapping so atomicMax(u32) == float max ----------
__device__ __forceinline__ unsigned f2ord(float f) {
    unsigned u = __float_as_uint(f);
    return (u & 0x80000000u) ? ~u : (u | 0x80000000u);
}
__device__ __forceinline__ float ord2f(unsigned u) {
    return __uint_as_float((u & 0x80000000u) ? (u ^ 0x80000000u) : ~u);
}

// =====================================================================
// f32 -> f16 streaming convert (n multiple of 4)
// =====================================================================
__global__ void gat_cvt_f16(const float* __restrict__ in,
                            _Float16* __restrict__ out, size_t n)
{
    size_t i = ((size_t)blockIdx.x * blockDim.x + threadIdx.x) * 4;
    if (i >= n) return;
    float4 v = *reinterpret_cast<const float4*>(in + i);
    v4h o;
    o[0] = (_Float16)v.x; o[1] = (_Float16)v.y;
    o[2] = (_Float16)v.z; o[3] = (_Float16)v.w;
    *reinterpret_cast<v4h*>(out + i) = o;
}

// =====================================================================
// Y[N x LDY] = Xh[N x K] * Wh^T  (Wh f16 [tiles*16 x K] row-major)
// f16 WMMA 16x16x32, f32 accumulate, software-pipelined (double-buffered
// fragments: iteration i+1 loads are issued before iteration i WMMAs).
// One wave -> 16 rows x (TM*16) cols.  Block (32, WAVES).
// Requires: N % 16 == 0, K % 32 == 0, weight panel padded to full tiles.
// A fragment (16-bit A 16x32, own order, not ISA table): lane half hl, K = hl*16 + i (contiguous 16 halves)
// B fragment: lane = column, K = hl*16 + i (contiguous 16 halves)
// C/D v8f: row = hl*8 + r, col = lane%16
// =====================================================================
template <int TM, int LDY, int K>
__global__ void gat_gemm_f16(const _Float16* __restrict__ Xh,
                             const _Float16* __restrict__ Wh,
                             float* __restrict__ Y)
{
    const int lane = threadIdx.x;            // 0..31
    const int hl   = lane >> 4;
    const int lm   = lane & 15;
    const int rowTile = blockIdx.x << 4;
    const int colBase = (threadIdx.y * TM) << 4;

    const _Float16* __restrict__ aptr = Xh + (size_t)(rowTile + lm) * K;
    const _Float16* __restrict__ wcol = Wh + (size_t)(colBase + lm) * K;

    auto loadA = [&](int k0) -> v16h {
        v8h x0 = *reinterpret_cast<const v8h*>(aptr + k0 + hl * 16);
        v8h x1 = *reinterpret_cast<const v8h*>(aptr + k0 + hl * 16 + 8);
        return __builtin_shufflevector(x0, x1, 0, 1, 2, 3, 4, 5, 6, 7,
                                               8, 9, 10, 11, 12, 13, 14, 15);
    };

    v8f acc[TM];
#pragma unroll
    for (int j = 0; j < TM; ++j) acc[j] = (v8f){};

    // prologue: fragments for k0 = 0
    v16h aCur = loadA(0);
    v16h bCur[TM];
#pragma unroll
    for (int j = 0; j < TM; ++j)
        bCur[j] = *reinterpret_cast<const v16h*>(wcol + (size_t)j * 16 * K + hl * 16);

#pragma unroll
    for (int i = 0; i < K / 32 - 1; ++i) {
        const int k1 = (i + 1) * 32;
        // issue next-iteration loads before consuming current fragments
        v16h aNxt = loadA(k1);
        v16h bNxt[TM];
#pragma unroll
        for (int j = 0; j < TM; ++j)
            bNxt[j] = *reinterpret_cast<const v16h*>(wcol + (size_t)j * 16 * K +
                                                     k1 + hl * 16);
#pragma unroll
        for (int j = 0; j < TM; ++j)
            acc[j] = __builtin_amdgcn_wmma_f32_16x16x32_f16(
                false, aCur, false, bCur[j], (short)0, acc[j], false, false);
        aCur = aNxt;
#pragma unroll
        for (int j = 0; j < TM; ++j) bCur[j] = bNxt[j];
    }
    // epilogue
#pragma unroll
    for (int j = 0; j < TM; ++j)
        acc[j] = __builtin_amdgcn_wmma_f32_16x16x32_f16(
            false, aCur, false, bCur[j], (short)0, acc[j], false, false);

    float* __restrict__ yp = Y + (size_t)(rowTile + hl * 8) * LDY + colBase + lm;
#pragma unroll
    for (int j = 0; j < TM; ++j)
#pragma unroll
        for (int r = 0; r < 8; ++r)
            yp[(size_t)r * LDY + j * 16] = acc[j][r];
}

// =====================================================================
// alpha_src[n][h] = sum_c h[n][h*C+c]*a_src[h][c]   (same for dst)
// one wave per node, wave-shuffle reduction; ldH = row stride of Hf
// =====================================================================
__global__ void gat_attn_coef(const float* __restrict__ Hf,
                              const float* __restrict__ Asv,
                              const float* __restrict__ Adv,
                              float* __restrict__ as_out,
                              float* __restrict__ ad_out,
                              int N, int Hh, int Cper, int ldH)
{
    const int wave = threadIdx.x >> 5;
    const int lane = threadIdx.x & 31;
    const int n = blockIdx.x * (blockDim.x >> 5) + wave;
    if (n >= N) return;  // wave-uniform
    const float* hrow = Hf + (size_t)n * ldH;
    for (int h = 0; h < Hh; ++h) {
        float ps = 0.f, pd = 0.f;
        for (int c = lane; c < Cper; c += 32) {
            float v = hrow[h * Cper + c];
            ps += v * Asv[h * Cper + c];
            pd += v * Adv[h * Cper + c];
        }
#pragma unroll
        for (int off = 16; off; off >>= 1) {
            ps += __shfl_xor(ps, off, 32);
            pd += __shfl_xor(pd, off, 32);
        }
        if (lane == 0) { as_out[n * Hh + h] = ps; ad_out[n * Hh + h] = pd; }
    }
}

// =====================================================================
// Edge pass 1: e = leaky_relu(as[src]+ad[dst]); segment-max via u32 atomicMax
// =====================================================================
__global__ void gat_edge_logits(const long long* __restrict__ esrc,
                                const long long* __restrict__ edst,
                                const float* __restrict__ as_,
                                const float* __restrict__ ad_,
                                float* __restrict__ elog,
                                unsigned* __restrict__ mkey,
                                int E, int Etot, int Hh)
{
    int e = blockIdx.x * blockDim.x + threadIdx.x;
    if (e >= Etot) return;
    long long s, d;
    if (e < E) { s = esrc[e]; d = edst[e]; } else { s = d = (long long)(e - E); }
    for (int h = 0; h < Hh; ++h) {
        float v = as_[s * Hh + h] + ad_[d * Hh + h];
        v = (v > 0.f) ? v : 0.2f * v;
        elog[(size_t)e * Hh + h] = v;
        atomicMax(&mkey[d * Hh + h], f2ord(v));
    }
}

// =====================================================================
// Edge pass 2: ex = exp(e - max[dst]); segment-sum denominator
// =====================================================================
__global__ void gat_edge_exp(const long long* __restrict__ edst,
                             float* __restrict__ elog,
                             const unsigned* __restrict__ mkey,
                             float* __restrict__ den,
                             int E, int Etot, int Hh)
{
    int e = blockIdx.x * blockDim.x + threadIdx.x;
    if (e >= Etot) return;
    long long d = (e < E) ? edst[e] : (long long)(e - E);
    for (int h = 0; h < Hh; ++h) {
        float m  = ord2f(mkey[d * Hh + h]);
        float ex = __expf(elog[(size_t)e * Hh + h] - m);
        elog[(size_t)e * Hh + h] = ex;
        atomicAdd(&den[d * Hh + h], ex);
    }
}

// =====================================================================
// Edge pass 3: agg[dst] += h[src] * (ex/den[dst]); 1 block = 1 edge
// ldH = row stride of Hf (may be padded); agg rows are compact (Hh*Cper)
// =====================================================================
__global__ void gat_edge_aggregate(const long long* __restrict__ esrc,
                                   const long long* __restrict__ edst,
                                   const float* __restrict__ Hf,
                                   const float* __restrict__ elog,
                                   const float* __restrict__ den,
                                   float* __restrict__ agg,
                                   int E, int Hh, int Cper, int ldH)
{
    const int e  = blockIdx.x;
    const int ch = threadIdx.x;
    const int HC = Hh * Cper;
    if (ch >= HC) return;
    long long s, d;
    if (e < E) { s = esrc[e]; d = edst[e]; } else { s = d = (long long)(e - E); }
    const int h = ch / Cper;
    float coef = elog[(size_t)e * Hh + h] / den[d * Hh + h];
    atomicAdd(&agg[(size_t)d * HC + ch], Hf[(size_t)s * ldH + ch] * coef);
}

// =====================================================================
// elementwise finalizers + fills
// =====================================================================
__global__ void gat_finalize_elu(float* __restrict__ buf,
                                 const float* __restrict__ b,
                                 size_t total, int HC)
{
    size_t i = (size_t)blockIdx.x * blockDim.x + threadIdx.x;
    if (i >= total) return;
    float v = buf[i] + b[i % HC];
    buf[i] = (v > 0.f) ? v : (__expf(v) - 1.f);
}

__global__ void gat_finalize_avg(const float* __restrict__ agg,
                                 const float* __restrict__ b,
                                 float* __restrict__ out,
                                 int N, int Hh, int NCc)
{
    int i = blockIdx.x * blockDim.x + threadIdx.x;
    if (i >= N * NCc) return;
    int n = i / NCc, c = i % NCc;
    float s = 0.f;
    for (int h = 0; h < Hh; ++h) s += agg[(size_t)n * Hh * NCc + h * NCc + c];
    float v = s / (float)Hh + b[c];
    out[i] = (v > 0.f) ? v : (__expf(v) - 1.f);
}

__global__ void gat_fill_f32(float* p, float v, size_t n) {
    size_t i = (size_t)blockIdx.x * blockDim.x + threadIdx.x;
    if (i < n) p[i] = v;
}
__global__ void gat_fill_u32(unsigned* p, unsigned v, size_t n) {
    size_t i = (size_t)blockIdx.x * blockDim.x + threadIdx.x;
    if (i < n) p[i] = v;
}

// =====================================================================
extern "C" void kernel_launch(void* const* d_in, const int* in_sizes, int n_in,
                              void* d_out, int out_size, void* d_ws, size_t ws_size,
                              hipStream_t stream)
{
    const float*     x   = (const float*)d_in[0];
    const long long* ei  = (const long long*)d_in[1];  // int64 edge_index [2,E]
    const float* W1 = (const float*)d_in[2];
    const float* as1 = (const float*)d_in[3];
    const float* ad1 = (const float*)d_in[4];
    const float* b1  = (const float*)d_in[5];
    const float* W2 = (const float*)d_in[6];
    const float* as2 = (const float*)d_in[7];
    const float* ad2 = (const float*)d_in[8];
    const float* b2  = (const float*)d_in[9];
    const float* W3 = (const float*)d_in[10];
    const float* as3 = (const float*)d_in[11];
    const float* ad3 = (const float*)d_in[12];
    const float* b3  = (const float*)d_in[13];
    float* out = (float*)d_out;

    const int FIN = 256, HC = 256, H = 4, C = 64, NC = 3, HNC = 12, HNCP = 16;
    const int N    = in_sizes[0] / FIN;   // 50000, divisible by 16
    const int E    = in_sizes[1] / 2;
    const int Etot = E + N;
    const long long* esrc = ei;
    const long long* edst = ei + E;

    // ---- workspace carve-out (256B aligned slices)
    char* wsp = (char*)d_ws;
    auto carve = [&](size_t bytes) -> void* {
        void* p = (void*)wsp;
        wsp += (bytes + 255) & ~(size_t)255;
        return p;
    };
    float*     hbuf  = (float*)    carve((size_t)N * HC * 4);    // GEMM output
    float*     xbuf  = (float*)    carve((size_t)N * HC * 4);    // agg / next input
    _Float16*  Xh    = (_Float16*) carve((size_t)N * HC * 2);    // f16 GEMM input
    _Float16*  Wh    = (_Float16*) carve((size_t)HC * HC * 2);   // f16 weight panel
    float*     asrc_ = (float*)    carve((size_t)N * H * 4);
    float*     adst_ = (float*)    carve((size_t)N * H * 4);
    unsigned*  mkey  = (unsigned*) carve((size_t)N * H * 4);
    float*     den   = (float*)    carve((size_t)N * H * 4);
    float*     elog  = (float*)    carve((size_t)Etot * H * 4);

    const int nrt = N / 16;
    const unsigned gNH   = (unsigned)(((size_t)N * H + 255) / 256);
    const unsigned gNHC  = (unsigned)(((size_t)N * HC + 255) / 256);
    const unsigned gNHNC = (unsigned)(((size_t)N * HNC + 255) / 256);
    const unsigned gEd   = (unsigned)((Etot + 255) / 256);
    const unsigned gCvtX = (unsigned)(((size_t)N * HC / 4 + 255) / 256);
    const unsigned gCvtW = (unsigned)(((size_t)HC * HC / 4 + 255) / 256);
    const unsigned gCvtW3= (unsigned)(((size_t)HNC * HC / 4 + 255) / 256);

    // =================== Layer 1 (in=256 -> H*C=256, concat) ===================
    gat_cvt_f16<<<gCvtX, 256, 0, stream>>>(x,  Xh, (size_t)N * HC);
    gat_cvt_f16<<<gCvtW, 256, 0, stream>>>(W1, Wh, (size_t)HC * HC);
    gat_gemm_f16<4, 256, 256><<<dim3(nrt, 1), dim3(32, 4), 0, stream>>>(Xh, Wh, hbuf);
    gat_attn_coef<<<(N + 7) / 8, 256, 0, stream>>>(hbuf, as1, ad1, asrc_, adst_, N, H, C, HC);
    gat_fill_u32<<<gNH, 256, 0, stream>>>(mkey, 0u, (size_t)N * H);
    gat_fill_f32<<<gNH, 256, 0, stream>>>(den, 0.f, (size_t)N * H);
    gat_fill_f32<<<gNHC, 256, 0, stream>>>(xbuf, 0.f, (size_t)N * HC);
    gat_edge_logits<<<gEd, 256, 0, stream>>>(esrc, edst, asrc_, adst_, elog, mkey, E, Etot, H);
    gat_edge_exp<<<gEd, 256, 0, stream>>>(edst, elog, mkey, den, E, Etot, H);
    gat_edge_aggregate<<<Etot, 256, 0, stream>>>(esrc, edst, hbuf, elog, den, xbuf, E, H, C, HC);
    gat_finalize_elu<<<gNHC, 256, 0, stream>>>(xbuf, b1, (size_t)N * HC, HC);

    // =================== Layer 2 (in=256 -> 256, concat) ===================
    gat_cvt_f16<<<gCvtX, 256, 0, stream>>>(xbuf, Xh, (size_t)N * HC);
    gat_cvt_f16<<<gCvtW, 256, 0, stream>>>(W2, Wh, (size_t)HC * HC);
    gat_gemm_f16<4, 256, 256><<<dim3(nrt, 1), dim3(32, 4), 0, stream>>>(Xh, Wh, hbuf);
    gat_attn_coef<<<(N + 7) / 8, 256, 0, stream>>>(hbuf, as2, ad2, asrc_, adst_, N, H, C, HC);
    gat_fill_u32<<<gNH, 256, 0, stream>>>(mkey, 0u, (size_t)N * H);
    gat_fill_f32<<<gNH, 256, 0, stream>>>(den, 0.f, (size_t)N * H);
    gat_fill_f32<<<gNHC, 256, 0, stream>>>(xbuf, 0.f, (size_t)N * HC);   // GEMM2 done reading
    gat_edge_logits<<<gEd, 256, 0, stream>>>(esrc, edst, asrc_, adst_, elog, mkey, E, Etot, H);
    gat_edge_exp<<<gEd, 256, 0, stream>>>(edst, elog, mkey, den, E, Etot, H);
    gat_edge_aggregate<<<Etot, 256, 0, stream>>>(esrc, edst, hbuf, elog, den, xbuf, E, H, C, HC);
    gat_finalize_elu<<<gNHC, 256, 0, stream>>>(xbuf, b2, (size_t)N * HC, HC);

    // =================== Layer 3 (in=256 -> H*NC=12, mean heads) ===================
    // weight panel padded to 16 rows (rows 12..15 zero) -> branch-free GEMM,
    // hbuf uses row stride 16 (cols 12..15 are don't-care)
    gat_cvt_f16<<<gCvtX, 256, 0, stream>>>(xbuf, Xh, (size_t)N * HC);
    gat_fill_u32<<<(unsigned)(((size_t)HNCP * HC * 2 / 4 + 255) / 256), 256, 0, stream>>>(
        (unsigned*)Wh, 0u, (size_t)HNCP * HC * 2 / 4);
    gat_cvt_f16<<<gCvtW3, 256, 0, stream>>>(W3, Wh, (size_t)HNC * HC);
    gat_gemm_f16<1, 16, 256><<<dim3(nrt, 1), dim3(32, 1), 0, stream>>>(Xh, Wh, hbuf);
    gat_attn_coef<<<(N + 7) / 8, 256, 0, stream>>>(hbuf, as3, ad3, asrc_, adst_, N, H, NC, HNCP);
    gat_fill_u32<<<gNH, 256, 0, stream>>>(mkey, 0u, (size_t)N * H);
    gat_fill_f32<<<gNH, 256, 0, stream>>>(den, 0.f, (size_t)N * H);
    gat_fill_f32<<<gNHNC, 256, 0, stream>>>(xbuf, 0.f, (size_t)N * HNC);
    gat_edge_logits<<<gEd, 256, 0, stream>>>(esrc, edst, asrc_, adst_, elog, mkey, E, Etot, H);
    gat_edge_exp<<<gEd, 256, 0, stream>>>(edst, elog, mkey, den, E, Etot, H);
    gat_edge_aggregate<<<Etot, 32, 0, stream>>>(esrc, edst, hbuf, elog, den, xbuf, E, H, NC, HNCP);
    gat_finalize_avg<<<(N * NC + 255) / 256, 256, 0, stream>>>(xbuf, b3, out, N, H, NC);
}
